// PointNetFeatureExtractPp_44659069944088
// MI455X (gfx1250) — compile-verified, matrix operand order checked by reading, not run on hardware
//
#include <hip/hip_runtime.h>
#include <hip/hip_bf16.h>
#include <math.h>

typedef __attribute__((ext_vector_type(16))) __bf16 v16bf;
typedef __attribute__((ext_vector_type(8)))  float  v8f;

#define BATCH 4
#define NPTS  8192
#define FXD   8
#define S1N   2048
#define S2N   512
#define KNN   64

// ---------------------------------------------------------------------------
// Branch-free tanh via hardware transcendentals (bf16-accuracy is plenty):
// tanh(x) = 1 - 2/(exp2(2*log2e*x) + 1)  -> v_exp_f32 + v_rcp_f32 + 3 VALU
// Saturates correctly: x >> 0 -> 1, x << 0 -> -1.
// ---------------------------------------------------------------------------
__device__ inline float fast_tanh(float x) {
  float e = __builtin_amdgcn_exp2f(x * 2.885390081777927f);  // 2*log2(e)
  return 1.0f - 2.0f * __builtin_amdgcn_rcpf(e + 1.0f);
}

// ---------------------------------------------------------------------------
// WMMA helpers: V_WMMA_F32_16X16X32_BF16, D = A(16x32) * B(32x16) + C
// Per-lane K pattern (ISA 7.12.2): two contiguous runs of 8 K-values:
//   k = (half<<3)+0..7   and   k = 16+(half<<3)+0..7,  half = lane>>4
// so both A (LDS, row-major per 16-row tile) and B (global, transposed
// weights Wt[n][kpad]) fragments are two 16-byte contiguous loads.
// ---------------------------------------------------------------------------
__device__ inline v8f wmma_bf16(v16bf a, v16bf b, v8f c) {
  return __builtin_amdgcn_wmma_f32_16x16x32_bf16(false, a, false, b,
                                                 (short)0, c, false, false);
}

__device__ inline v16bf load_a_frag(const __bf16* buf, int stride, int lane, int kt) {
  const __bf16* p = buf + (lane & 15) * stride + kt * 32 + ((lane >> 4) << 3);
  v16bf r;
#pragma unroll
  for (int e = 0; e < 8; ++e) { r[e] = p[e]; r[8 + e] = p[16 + e]; }
  return r;
}

__device__ inline v16bf load_b_frag(const __bf16* Wt, int kpad, int lane, int kt, int nt) {
  const __bf16* p = Wt + (size_t)(nt * 16 + (lane & 15)) * kpad + kt * 32 + ((lane >> 4) << 3);
  v16bf r;
#pragma unroll
  for (int e = 0; e < 8; ++e) { r[e] = p[e]; r[8 + e] = p[16 + e]; }
  return r;
}

// One Linear+Tanh layer on a 16-row tile: LDS(bf16) -> LDS(bf16).
// Compile-time dims: A-fragments preloaded to registers, reused over N-tiles.
template <int KPAD, int NOUT>
__device__ inline void mlp_layer(const __bf16* in, const __bf16* Wt,
                                 const float* bias, __bf16* outb,
                                 int ostride, int lane) {
  constexpr int KTC = KPAD / 32, NTC = NOUT / 16;
  v16bf afr[KTC];
#pragma unroll
  for (int kt = 0; kt < KTC; ++kt) afr[kt] = load_a_frag(in, KPAD, lane, kt);
  for (int nt = 0; nt < NTC; ++nt) {
    v8f acc = {};
#pragma unroll
    for (int kt = 0; kt < KTC; ++kt)
      acc = wmma_bf16(afr[kt], load_b_frag(Wt, KPAD, lane, kt, nt), acc);
    float bs = bias[nt * 16 + (lane & 15)];
#pragma unroll
    for (int v = 0; v < 8; ++v) {
      int m = v + ((lane >> 4) << 3);                 // C/D layout: vgpr v -> row
      outb[m * ostride + nt * 16 + (lane & 15)] = (__bf16)fast_tanh(acc[v] + bs);
    }
  }
}

__device__ inline float atomicMaxF32(float* addr, float val) {
  int* ai = (int*)addr;
  int old = __float_as_int(*addr);
  while (__int_as_float(old) < val) {
    int assumed = old;
    old = atomicCAS(ai, assumed, __float_as_int(val));
    if (old == assumed) break;
  }
  return __int_as_float(old);
}

// ---------------------------------------------------------------------------
// Weight prep: f32 W[a][b]  ->  bf16 transposed + K-padded Wt[b][kpad]
// ---------------------------------------------------------------------------
__global__ void convert_w_kernel(const float* __restrict__ W, __bf16* __restrict__ dst,
                                 int a, int b, int kpad) {
  int i = blockIdx.x * 256 + threadIdx.x;
  if (i >= kpad * b) return;
  int n = i / kpad, k = i % kpad;
  dst[i] = (k < a) ? (__bf16)W[k * b + n] : (__bf16)0.0f;
}

__global__ void init_g_kernel(float* g) {
  int i = blockIdx.x * 256 + threadIdx.x;
  if (i < BATCH * 1024) g[i] = -1e30f;
}

// ---------------------------------------------------------------------------
// local = tanh-MLP(pos): 2(pad32) -> 64 -> 64 -> 128, rows = B*N
// block = 128 (4 waves), each wave owns one 16-row tile
// ---------------------------------------------------------------------------
__global__ __launch_bounds__(128)
void local_mlp_kernel(const float* __restrict__ pos,
                      const __bf16* W1t, const float* b1,
                      const __bf16* W2t, const float* b2,
                      const __bf16* W3t, const float* b3,
                      float* __restrict__ out) {
  __shared__ __align__(16) __bf16 bufA[4][16 * 64];
  __shared__ __align__(16) __bf16 bufB[4][16 * 64];
  const int t = threadIdx.x, wave = t >> 5, lane = t & 31;
  const size_t row0 = (size_t)blockIdx.x * 64 + wave * 16;

  for (int e = lane; e < 16 * 32; e += 32) {            // input tile 16x32 (zero-pad)
    int rr = e / 32, c = e % 32;
    float v = (c < 2) ? pos[(row0 + rr) * 2 + c] : 0.0f;
    bufA[wave][rr * 32 + c] = (__bf16)v;
  }
  mlp_layer<32, 64>(bufA[wave], W1t, b1, bufB[wave], 64, lane);
  mlp_layer<64, 64>(bufB[wave], W2t, b2, bufA[wave], 64, lane);
  {                                                     // final 64 -> 128, store f32
    v16bf afr[2];
#pragma unroll
    for (int kt = 0; kt < 2; ++kt) afr[kt] = load_a_frag(bufA[wave], 64, lane, kt);
    for (int nt = 0; nt < 8; ++nt) {
      v8f acc = {};
#pragma unroll
      for (int kt = 0; kt < 2; ++kt)
        acc = wmma_bf16(afr[kt], load_b_frag(W3t, 64, lane, kt, nt), acc);
      float bs = b3[nt * 16 + (lane & 15)];
#pragma unroll
      for (int v = 0; v < 8; ++v) {
        int m = v + ((lane >> 4) << 3);
        out[(row0 + m) * 128 + nt * 16 + (lane & 15)] = fast_tanh(acc[v] + bs);
      }
    }
  }
}

// ---------------------------------------------------------------------------
// Farthest point sampling: 1 workgroup per batch, mind2 resident in LDS.
// Writes sampled positions posq[b*s + i]. Deterministic start at index 0.
// ---------------------------------------------------------------------------
__global__ __launch_bounds__(256)
void fps_kernel(const float* __restrict__ pos, int n, int s,
                float* __restrict__ posq) {
  __shared__ float mind2[NPTS];
  __shared__ float rmax[256];
  __shared__ int   rarg[256];
  __shared__ int   s_last;
  const int b = blockIdx.x, t = threadIdx.x;
  const float* p = pos + (size_t)b * n * 2;
  for (int i = t; i < n; i += 256) mind2[i] = 1e30f;
  if (t == 0) {
    posq[(size_t)(b * s) * 2 + 0] = p[0];
    posq[(size_t)(b * s) * 2 + 1] = p[1];
    s_last = 0;
  }
  __syncthreads();
  int last = 0;
  for (int it = 1; it < s; ++it) {
    float lx = p[last * 2], ly = p[last * 2 + 1];
    float bm = -1.0f; int ba = 0;
    for (int i = t; i < n; i += 256) {
      float dx = p[i * 2] - lx, dy = p[i * 2 + 1] - ly;
      float m = fminf(mind2[i], dx * dx + dy * dy);
      mind2[i] = m;
      if (m > bm) { bm = m; ba = i; }
    }
    rmax[t] = bm; rarg[t] = ba;
    __syncthreads();
    for (int o = 128; o > 0; o >>= 1) {
      if (t < o && rmax[t + o] > rmax[t]) { rmax[t] = rmax[t + o]; rarg[t] = rarg[t + o]; }
      __syncthreads();
    }
    if (t == 0) {
      s_last = rarg[0];
      posq[(size_t)(b * s + it) * 2 + 0] = p[s_last * 2];
      posq[(size_t)(b * s + it) * 2 + 1] = p[s_last * 2 + 1];
    }
    __syncthreads();
    last = s_last;
  }
}

// ---------------------------------------------------------------------------
// Exact 64-NN per query: d2 array in LDS, 64 rounds of block argmin w/ removal
// 1 workgroup (256 thr) per query sample.
// ---------------------------------------------------------------------------
__global__ __launch_bounds__(256)
void knn_kernel(const float* __restrict__ pos_src, const float* __restrict__ posq,
                int n, int S, int* __restrict__ nbr, float* __restrict__ nd2) {
  __shared__ float d2s[NPTS];
  __shared__ float rmin[256];
  __shared__ int   rarg[256];
  const int q = blockIdx.x, b = q / S, t = threadIdx.x;
  const float* p = pos_src + (size_t)b * n * 2;
  const float qx = posq[(size_t)q * 2], qy = posq[(size_t)q * 2 + 1];
  for (int i = t; i < n; i += 256) {
    float dx = p[i * 2] - qx, dy = p[i * 2 + 1] - qy;
    d2s[i] = dx * dx + dy * dy;
  }
  __syncthreads();
  for (int j = 0; j < KNN; ++j) {
    float bm = 1e31f; int ba = 0;
    for (int i = t; i < n; i += 256) {
      float v = d2s[i];
      if (v < bm) { bm = v; ba = i; }
    }
    rmin[t] = bm; rarg[t] = ba;
    __syncthreads();
    for (int o = 128; o > 0; o >>= 1) {
      if (t < o && rmin[t + o] < rmin[t]) { rmin[t] = rmin[t + o]; rarg[t] = rarg[t + o]; }
      __syncthreads();
    }
    if (t == 0) {
      nbr[(size_t)q * KNN + j] = rarg[0];
      nd2[(size_t)q * KNN + j] = rmin[0];
      d2s[rarg[0]] = 1e32f;
    }
    __syncthreads();
  }
}

// ---------------------------------------------------------------------------
// Set-abstraction MLP: gather 64 neighbor rows, 3 tanh layers (WMMA),
// radius-masked max over neighbors fused into last layer.
// block = 64 (2 waves), 1 workgroup per sample; each wave does 2 row tiles.
// Compile-time dims via template; A-fragments register-resident.
// ---------------------------------------------------------------------------
template <int K0PAD, int W1, int W2, int W3, int FDIM, bool PREPEND>
__global__ __launch_bounds__(64)
void sa_mlp_kernel(const float* __restrict__ feat, int n_src,
                   const float* __restrict__ pos_src, const float* __restrict__ posq,
                   const int* __restrict__ nbr, const float* __restrict__ nd2,
                   const __bf16* W1t, const float* b1,
                   const __bf16* W2t, const float* b2,
                   const __bf16* W3t, const float* b3,
                   float r2, int S, float* __restrict__ out) {
  __shared__ __align__(16) __bf16 bufA[2][16 * 256];
  __shared__ __align__(16) __bf16 bufB[2][16 * 256];
  __shared__ float red[8][256];
  const int q = blockIdx.x, b = q / S;
  const int t = threadIdx.x, wave = t >> 5, lane = t & 31;
  const float* ps = pos_src + (size_t)b * n_src * 2;
  const float qx = posq[(size_t)q * 2], qy = posq[(size_t)q * 2 + 1];
  constexpr int base = PREPEND ? 2 : 0;

  for (int tile = wave; tile < 4; tile += 2) {
    for (int e = lane; e < 16 * K0PAD; e += 32) {       // gather input tile
      int rr = e / K0PAD, c = e % K0PAD;
      int nidx = nbr[(size_t)q * KNN + tile * 16 + rr];
      float v = 0.0f;
      if (PREPEND && c < 2)                  v = ps[nidx * 2 + c];
      else if (c >= base && c < base + FDIM) v = feat[((size_t)b * n_src + nidx) * FDIM + (c - base)];
      else if (c == base + FDIM)             v = ps[nidx * 2 + 0] - qx;
      else if (c == base + FDIM + 1)         v = ps[nidx * 2 + 1] - qy;
      bufA[wave][rr * K0PAD + c] = (__bf16)v;
    }
    mlp_layer<K0PAD, W1>(bufA[wave], W1t, b1, bufB[wave], W1, lane);
    mlp_layer<W1, W2>(bufB[wave], W2t, b2, bufA[wave], W2, lane);
    // last layer + radius-masked neighbor max
    constexpr int NTC = W3 / 16, KTC = W2 / 32;
    v16bf afr[KTC];
#pragma unroll
    for (int kt = 0; kt < KTC; ++kt) afr[kt] = load_a_frag(bufA[wave], W2, lane, kt);
    for (int nt = 0; nt < NTC; ++nt) {
      v8f acc = {};
#pragma unroll
      for (int kt = 0; kt < KTC; ++kt)
        acc = wmma_bf16(afr[kt], load_b_frag(W3t, W2, lane, kt, nt), acc);
      float bs = b3[nt * 16 + (lane & 15)];
      float mx = -1e30f;
#pragma unroll
      for (int v = 0; v < 8; ++v) {
        int m = v + ((lane >> 4) << 3);
        float d2v = nd2[(size_t)q * KNN + tile * 16 + m];
        float val = fast_tanh(acc[v] + bs);
        if (d2v <= r2) mx = fmaxf(mx, val);
      }
      red[tile * 2 + (lane >> 4)][nt * 16 + (lane & 15)] = mx;
    }
  }
  __syncthreads();
  for (int c = t; c < W3; c += 64) {
    float m = red[0][c];
#pragma unroll
    for (int p8 = 1; p8 < 8; ++p8) m = fmaxf(m, red[p8][c]);
    out[(size_t)q * W3 + c] = m;
  }
}

// ---------------------------------------------------------------------------
// Global MLP: 258(pad288) -> 256 -> 512 -> 1024, fused per-cloud max (atomic).
// block = 32 (1 wave) per 16-row tile; 128 tiles total.
// ---------------------------------------------------------------------------
__global__ __launch_bounds__(32)
void global_mlp_kernel(const float* __restrict__ h2, const float* __restrict__ pos2,
                       const __bf16* W1t, const float* b1,
                       const __bf16* W2t, const float* b2,
                       const __bf16* W3t, const float* b3,
                       float* __restrict__ gout) {
  __shared__ __align__(16) __bf16 bufA[16 * 512];
  __shared__ __align__(16) __bf16 bufB[16 * 512];
  __shared__ float pr[32];
  const int lane = threadIdx.x;
  const int row0 = blockIdx.x * 16;
  const int b = row0 / S2N;

  for (int e = lane; e < 16 * 288; e += 32) {
    int rr = e / 288, c = e % 288;
    size_t r = (size_t)row0 + rr;
    float v = 0.0f;
    if (c < 256)      v = h2[r * 256 + c];
    else if (c < 258) v = pos2[r * 2 + (c - 256)];
    bufA[rr * 288 + c] = (__bf16)v;
  }
  mlp_layer<288, 256>(bufA, W1t, b1, bufB, 256, lane);
  mlp_layer<256, 512>(bufB, W2t, b2, bufA, 512, lane);
  for (int nt = 0; nt < 64; ++nt) {                     // 512 -> 1024 + row max
    v8f acc = {};
    for (int kt = 0; kt < 16; ++kt) {                   // KTC=16: reload A (VGPR budget)
      v16bf a = load_a_frag(bufA, 512, lane, kt);
      v16bf w = load_b_frag(W3t, 512, lane, kt, nt);
      acc = wmma_bf16(a, w, acc);
    }
    float bs = b3[nt * 16 + (lane & 15)];
    float mx = -1e30f;
#pragma unroll
    for (int v = 0; v < 8; ++v) mx = fmaxf(mx, fast_tanh(acc[v] + bs));
    pr[lane] = mx;
    __syncthreads();
    if (lane < 16)
      atomicMaxF32(&gout[(size_t)b * 1024 + nt * 16 + lane], fmaxf(pr[lane], pr[lane + 16]));
    __syncthreads();
  }
}

// ---------------------------------------------------------------------------
extern "C" void kernel_launch(void* const* d_in, const int* in_sizes, int n_in,
                              void* d_out, int out_size, void* d_ws, size_t ws_size,
                              hipStream_t stream) {
  (void)in_sizes; (void)n_in; (void)out_size; (void)ws_size;
  const float* x   = (const float*)d_in[0];
  const float* pos = (const float*)d_in[1];

  char* ws = (char*)d_ws; size_t off = 0;
  auto take = [&](size_t bytes) -> void* {
    void* p = ws + off; off = (off + bytes + 255) & ~(size_t)255; return p;
  };
  // transposed bf16 weights: Wt[nout][kpad]
  __bf16* lw1  = (__bf16*)take((size_t)64  * 32  * 2);
  __bf16* lw2  = (__bf16*)take((size_t)64  * 64  * 2);
  __bf16* lw3  = (__bf16*)take((size_t)128 * 64  * 2);
  __bf16* s1w1 = (__bf16*)take((size_t)64  * 32  * 2);
  __bf16* s1w2 = (__bf16*)take((size_t)64  * 64  * 2);
  __bf16* s1w3 = (__bf16*)take((size_t)128 * 64  * 2);
  __bf16* s2w1 = (__bf16*)take((size_t)128 * 160 * 2);
  __bf16* s2w2 = (__bf16*)take((size_t)128 * 128 * 2);
  __bf16* s2w3 = (__bf16*)take((size_t)256 * 128 * 2);
  __bf16* gw1  = (__bf16*)take((size_t)256 * 288 * 2);
  __bf16* gw2  = (__bf16*)take((size_t)512 * 256 * 2);
  __bf16* gw3  = (__bf16*)take((size_t)1024 * 512 * 2);
  float* pos1  = (float*)take((size_t)BATCH * S1N * 2 * 4);
  int*   nbr1  = (int*)  take((size_t)BATCH * S1N * KNN * 4);
  float* nd2_1 = (float*)take((size_t)BATCH * S1N * KNN * 4);
  float* h1    = (float*)take((size_t)BATCH * S1N * 128 * 4);
  float* pos2  = (float*)take((size_t)BATCH * S2N * 2 * 4);
  int*   nbr2  = (int*)  take((size_t)BATCH * S2N * KNN * 4);
  float* nd2_2 = (float*)take((size_t)BATCH * S2N * KNN * 4);
  float* h2    = (float*)take((size_t)BATCH * S2N * 256 * 4);

  auto cw = [&](int widx, __bf16* dst, int a, int b, int kpad) {
    int total = kpad * b;
    convert_w_kernel<<<(total + 255) / 256, 256, 0, stream>>>(
        (const float*)d_in[widx], dst, a, b, kpad);
  };
  cw(2,  lw1,  2,   64,   32);  cw(4,  lw2,  64,  64,  64);  cw(6,  lw3,  64,  128, 64);
  cw(8,  s1w1, 12,  64,   32);  cw(10, s1w2, 64,  64,  64);  cw(12, s1w3, 64,  128, 64);
  cw(14, s2w1, 130, 128, 160);  cw(16, s2w2, 128, 128, 128); cw(18, s2w3, 128, 256, 128);
  cw(20, gw1,  258, 256, 288);  cw(22, gw2,  256, 512, 256); cw(24, gw3,  512, 1024, 512);

  float* out_local = (float*)d_out;                           // (B,N,128)
  float* out_g     = out_local + (size_t)BATCH * NPTS * 128;  // (B,1024)

  // local MLP on pos
  local_mlp_kernel<<<(BATCH * NPTS) / 64, 128, 0, stream>>>(
      pos, lw1, (const float*)d_in[3], lw2, (const float*)d_in[5],
      lw3, (const float*)d_in[7], out_local);

  // SA1
  fps_kernel<<<BATCH, 256, 0, stream>>>(pos, NPTS, S1N, pos1);
  knn_kernel<<<BATCH * S1N, 256, 0, stream>>>(pos, pos1, NPTS, S1N, nbr1, nd2_1);
  sa_mlp_kernel<32, 64, 64, 128, FXD, true><<<BATCH * S1N, 64, 0, stream>>>(
      x, NPTS, pos, pos1, nbr1, nd2_1,
      s1w1, (const float*)d_in[9], s1w2, (const float*)d_in[11],
      s1w3, (const float*)d_in[13],
      0.08f * 0.08f, S1N, h1);

  // SA2
  fps_kernel<<<BATCH, 256, 0, stream>>>(pos1, S1N, S2N, pos2);
  knn_kernel<<<BATCH * S2N, 256, 0, stream>>>(pos1, pos2, S1N, S2N, nbr2, nd2_2);
  sa_mlp_kernel<160, 128, 128, 256, 128, false><<<BATCH * S2N, 64, 0, stream>>>(
      h1, S1N, pos1, pos2, nbr2, nd2_2,
      s2w1, (const float*)d_in[15], s2w2, (const float*)d_in[17],
      s2w3, (const float*)d_in[19],
      0.25f * 0.25f, S2N, h2);

  // global MLP + per-cloud max
  init_g_kernel<<<(BATCH * 1024 + 255) / 256, 256, 0, stream>>>(out_g);
  global_mlp_kernel<<<(BATCH * S2N) / 16, 32, 0, stream>>>(
      h2, pos2, gw1, (const float*)d_in[21], gw2, (const float*)d_in[23],
      gw3, (const float*)d_in[25], out_g);
}
